// MixedMultiHeadAttention_89292370084211
// MI455X (gfx1250) — matrix operand order repaired and checked by reading, not compiled, not run on hardware
//
#include <hip/hip_runtime.h>

typedef __bf16 bf16;
typedef __attribute__((ext_vector_type(16))) __bf16 v16bf;
typedef __attribute__((ext_vector_type(8)))  __bf16 bf16x8;
typedef __attribute__((ext_vector_type(8)))  float  v8f;
typedef __attribute__((ext_vector_type(4)))  int    int4v;

constexpr int Bc = 2, Tc = 2048, Dc = 512, Hc = 8, DKc = 64;
constexpr int NROW = Bc * Tc;       // 4096 rows of the (b,t) GEMMs
constexpr int NQKV = 3 * Hc * DKc;  // 1536 fused projection columns
constexpr float NEG_BIG = -3.0e38f;

// ---------------- CDNA5 async global->LDS staging (guarded) ----------------
#if defined(__has_builtin)
#if __has_builtin(__builtin_amdgcn_global_load_async_to_lds_b128) && \
    __has_builtin(__builtin_amdgcn_s_wait_asynccnt)
#define USE_ASYNC_LDS 1
#endif
#endif
#ifndef USE_ASYNC_LDS
#define USE_ASYNC_LDS 0
#endif

__device__ __forceinline__ void copy16_to_lds(const bf16* g, bf16* l) {
#if USE_ASYNC_LDS
  // Builtin signature (from compiler diagnostic): param0 = global int4*,
  // param1 = LDS int4*, then imm offset, imm cpol.
  __builtin_amdgcn_global_load_async_to_lds_b128(
      (__attribute__((address_space(1))) int4v*)g,
      (__attribute__((address_space(3))) int4v*)l, 0, 0);
#else
  *(bf16x8*)l = *(const bf16x8*)g;
#endif
}

template <int N>
__device__ __forceinline__ void async_wait() {
#if USE_ASYNC_LDS
  __builtin_amdgcn_s_wait_asynccnt(N);
#endif
}

// ---------------- WMMA helpers (CDNA5 layouts, wave32) ----------------

__device__ __forceinline__ v8f wmma_bf16(v16bf a, v16bf b, v8f c) {
  // D(f32 16x16) = A(bf16 16x32) * B(bf16 32x16) + C
  return __builtin_amdgcn_wmma_f32_16x16x32_bf16(false, a, false, b, (short)0, c,
                                                 false, false);
}

// A fragment: lane (m = lane&15, half = lane>>4); elem e -> K = half*16+e.
// Source is row-major [16][ld] bf16.
__device__ __forceinline__ v16bf load_a_frag(const bf16* base, int ld, int lane) {
  int m = lane & 15, half = lane >> 4;
  const bf16* p = base + m * ld + half * 16;
  bf16x8 lo = *(const bf16x8*)p;         // K = half*16 .. +7
  bf16x8 hi = *(const bf16x8*)(p + 8);   // K = half*16+8 .. +15
  v16bf r;
#pragma unroll
  for (int i = 0; i < 8; ++i) { r[i] = lo[i]; r[i + 8] = hi[i]; }
  return r;
}

// B fragment: lane (n = lane&15, half = lane>>4); elem e -> K = half*16+e.
// Source layout: base[n*ld + k] == B[k][n] (K-contiguous per column).
__device__ __forceinline__ v16bf load_b_frag(const bf16* base, int ld, int lane) {
  int n = lane & 15, half = lane >> 4;
  return *(const v16bf*)(base + n * ld + half * 16);
}

// Reductions across the 16 lanes of a half-wave (C-layout row reductions).
__device__ __forceinline__ float rowmax16(float v) {
#pragma unroll
  for (int m = 1; m < 16; m <<= 1) v = fmaxf(v, __shfl_xor(v, m, 32));
  return v;
}
__device__ __forceinline__ float rowsum16(float v) {
#pragma unroll
  for (int m = 1; m < 16; m <<= 1) v += __shfl_xor(v, m, 32);
  return v;
}

// ---------------- Kernel 1: fp32 -> bf16 packing / transposes ----------------
__global__ void convert_pack(const float* __restrict__ q, const float* __restrict__ v,
                             const float* __restrict__ Wq, const float* __restrict__ Wk,
                             const float* __restrict__ Wv, const float* __restrict__ Wo,
                             bf16* __restrict__ qbf, bf16* __restrict__ vbf,
                             bf16* __restrict__ wqkvT, bf16* __restrict__ woT) {
  int i = blockIdx.x * blockDim.x + threadIdx.x;
  if (i < NROW * Dc) {
    qbf[i] = (bf16)q[i];
    vbf[i] = (bf16)v[i];
  }
  if (i < NQKV * Dc) {
    int row = i >> 9, d = i & 511;          // row = n, col = d
    const float* W = (row < 512) ? Wq : (row < 1024 ? Wk : Wv);
    int nn = row & 511;
    int h = nn >> 6, k = nn & 63;
    wqkvT[i] = (bf16)W[(h * Dc + d) * DKc + k];
  }
  if (i < Dc * Dc) {
    int d = i >> 9, nn = i & 511;           // Wo flat: (h*64+k)*512 + d
    woT[i] = (bf16)Wo[nn * Dc + d];
  }
}

// ---------------- Kernel 2: fused QKV projection GEMM ----------------
__global__ void proj_gemm(const bf16* __restrict__ qbf, const bf16* __restrict__ vbf,
                          const bf16* __restrict__ wqkvT,
                          bf16* __restrict__ Qb, bf16* __restrict__ Kb,
                          bf16* __restrict__ Vtb) {
  int w = threadIdx.x >> 5, lane = threadIdx.x & 31;
  int mi = w & 3, ni = w >> 2;
  int m0 = blockIdx.x * 64 + mi * 16;
  int n0 = blockIdx.y * 32 + ni * 16;
  const bf16* A = (n0 < 512) ? qbf : vbf;   // q-proj uses query; k/v use value
  const bf16* Arow = A + m0 * Dc;
  const bf16* Brow = wqkvT + n0 * Dc;
  v8f acc = {};
#pragma unroll 4
  for (int c = 0; c < Dc; c += 32) {
    v16bf a = load_a_frag(Arow + c, Dc, lane);
    v16bf b = load_b_frag(Brow + c, Dc, lane);
    acc = wmma_bf16(a, b, acc);
  }
  int half = lane >> 4, nc = lane & 15;
  int n = n0 + nc;
#pragma unroll
  for (int r = 0; r < 8; ++r) {
    int row = m0 + r + half * 8;
    int b = row >> 11, t = row & (Tc - 1);
    bf16 val = (bf16)acc[r];
    if (n < 512) {
      int h = n >> 6, k = n & 63;
      Qb[((b * Hc + h) * Tc + t) * DKc + k] = val;
    } else if (n < 1024) {
      int nn = n - 512, h = nn >> 6, k = nn & 63;
      Kb[((b * Hc + h) * Tc + t) * DKc + k] = val;
    } else {
      int nn = n - 1024, h = nn >> 6, k = nn & 63;
      Vtb[((b * Hc + h) * DKc + k) * Tc + t] = val;
    }
  }
}

// ---------------- Kernel 3: flash attention with async LDS staging ----------
// Block = 8 waves = 128 t-rows of one (b,h); K/V s-tiles are staged into LDS
// once per block (double-buffered, ASYNCcnt-tracked) and shared by all waves.
__global__ void attn_flash(const bf16* __restrict__ Qb, const bf16* __restrict__ Kb,
                           const bf16* __restrict__ Vtb,
                           float* __restrict__ Ows, float* __restrict__ mst,
                           float* __restrict__ lst, const int* __restrict__ lsp) {
  int tid = threadIdx.x;
  int w = tid >> 5, lane = tid & 31;
  int half = lane >> 4, nc = lane & 15;
  int h = blockIdx.y, b = blockIdx.z;
  int t0 = blockIdx.x * 128 + w * 16;
  int hg = h >> 1;                 // 0=global 1=local 2=forward 3=backward
  int ls = *lsp;

  const bf16* Qp = Qb + ((size_t)(b * Hc + h) * Tc + t0) * DKc;
  const bf16* Kbase = Kb + (size_t)(b * Hc + h) * Tc * DKc;
  const bf16* Vbase = Vtb + (size_t)(b * Hc + h) * DKc * Tc;

  v16bf aq0 = load_a_frag(Qp, DKc, lane);        // k = 0..31
  v16bf aq1 = load_a_frag(Qp + 32, DKc, lane);   // k = 32..63

  float m[8], l[8];
  v8f o[4] = {{}, {}, {}, {}};
#pragma unroll
  for (int r = 0; r < 8; ++r) { m[r] = NEG_BIG; l[r] = 0.0f; }

  __shared__ __align__(32) bf16 Ktile[2][32 * 64];  // [s][k]
  __shared__ __align__(32) bf16 Vtile[2][64 * 32];  // [k][s]
  __shared__ __align__(32) bf16 Plds[8][16 * 32];
  bf16* Pw = &Plds[w][0];

  // One b128 of K and one of V per thread per stage (4KB + 4KB per tile).
  auto stage = [&](int buf, int s0) {
    copy16_to_lds(Kbase + (size_t)s0 * DKc + tid * 8, &Ktile[buf][tid * 8]);
    copy16_to_lds(Vbase + (size_t)(tid >> 2) * Tc + s0 + (tid & 3) * 8,
                  &Vtile[buf][(tid >> 2) * 32 + (tid & 3) * 8]);
  };

  constexpr int NIT = Tc / 32;  // 64 s-tiles
  stage(0, 0);
  for (int it = 0; it < NIT; ++it) {
    int cur = it & 1;
    int s0 = it * 32;
    if (it + 1 < NIT) {
      stage(cur ^ 1, s0 + 32);  // prefetch next tile (2 more async ops)
      async_wait<2>();          // wait only for the current tile's 2 ops
    } else {
      async_wait<0>();
    }
    __syncthreads();            // staged data visible to all waves

    const bf16* Kt = &Ktile[cur][0];
    const bf16* Vt = &Vtile[cur][0];

    // ---- scores: two 16x16 tiles over s-columns [s0, s0+32) ----
    v8f S0 = {}, S1 = {};
    S0 = wmma_bf16(aq0, load_b_frag(Kt, 64, lane), S0);
    S0 = wmma_bf16(aq1, load_b_frag(Kt + 32, 64, lane), S0);
    S1 = wmma_bf16(aq0, load_b_frag(Kt + 16 * 64, 64, lane), S1);
    S1 = wmma_bf16(aq1, load_b_frag(Kt + 16 * 64 + 32, 64, lane), S1);

    // ---- mask + online softmax per row ----
    int sA = s0 + nc, sB = s0 + 16 + nc;
#pragma unroll
    for (int r = 0; r < 8; ++r) {
      int t = t0 + r + half * 8;
      float x0 = S0[r] * 0.125f;     // 1/sqrt(64)
      float x1 = S1[r] * 0.125f;
      bool k0, k1;
      if (hg == 0)      { k0 = true;               k1 = true; }
      else if (hg == 1) { int d0 = t - sA, d1 = t - sB;
                          k0 = (d0 <= ls) && (-d0 <= ls);
                          k1 = (d1 <= ls) && (-d1 <= ls); }
      else if (hg == 2) { k0 = (sA >= t);          k1 = (sB >= t); }
      else              { k0 = (sA <= t);          k1 = (sB <= t); }
      x0 = k0 ? x0 : NEG_BIG;
      x1 = k1 ? x1 : NEG_BIG;

      float rmax = rowmax16(fmaxf(x0, x1));
      float mn = fmaxf(m[r], rmax);
      float alpha = __expf(m[r] - mn);
      float p0 = __expf(x0 - mn);
      float p1 = __expf(x1 - mn);
      l[r] = l[r] * alpha + rowsum16(p0 + p1);
      m[r] = mn;
#pragma unroll
      for (int j = 0; j < 4; ++j) o[j][r] *= alpha;
      // C-layout -> row-major P tile in LDS (for A-fragment reload)
      Pw[(r + half * 8) * 32 + nc] = (bf16)p0;
      Pw[(r + half * 8) * 32 + 16 + nc] = (bf16)p1;
    }
    __builtin_amdgcn_wave_barrier();

    // ---- O += P(16x32) * V(32x64) ----
    v16bf ap = load_a_frag(Pw, 32, lane);
#pragma unroll
    for (int j = 0; j < 4; ++j) {
      o[j] = wmma_bf16(ap, load_b_frag(Vt + 16 * j * 32, 32, lane), o[j]);
    }
    __syncthreads();            // all waves done with `cur` before restaging it
  }

  // ---- write row-normalized O and per-row stats ----
  float* Orow = Ows + ((size_t)(b * Hc + h) * Tc + t0) * DKc;
#pragma unroll
  for (int r = 0; r < 8; ++r) {
    int trow = r + half * 8;
    float inv = 1.0f / l[r];
#pragma unroll
    for (int j = 0; j < 4; ++j)
      Orow[trow * DKc + 16 * j + nc] = o[j][r] * inv;
    if (nc == 0) {
      mst[(size_t)(b * Hc + h) * Tc + t0 + trow] = m[r];
      lst[(size_t)(b * Hc + h) * Tc + t0 + trow] = l[r];
    }
  }
}

// ---------------- Kernel 4: joint-softmax correction + rescale ----------------
__global__ void stats_rescale(const float* __restrict__ Ows,
                              const float* __restrict__ mst,
                              const float* __restrict__ lst,
                              bf16* __restrict__ attbf) {
  int bh = blockIdx.x;
  int b = bh >> 3, h = bh & 7;
  __shared__ float red[256];
  int tid = threadIdx.x;

  float lm = NEG_BIG;
  for (int t = tid; t < Tc; t += 256) lm = fmaxf(lm, mst[(size_t)bh * Tc + t]);
  red[tid] = lm;
  __syncthreads();
  for (int s = 128; s > 0; s >>= 1) {
    if (tid < s) red[tid] = fmaxf(red[tid], red[tid + s]);
    __syncthreads();
  }
  float M = red[0];
  __syncthreads();

  float lz = 0.0f;
  for (int t = tid; t < Tc; t += 256)
    lz += lst[(size_t)bh * Tc + t] * __expf(mst[(size_t)bh * Tc + t] - M);
  red[tid] = lz;
  __syncthreads();
  for (int s = 128; s > 0; s >>= 1) {
    if (tid < s) red[tid] += red[tid + s];
    __syncthreads();
  }
  float invZ = 1.0f / red[0];

  for (int idx = tid; idx < Tc * DKc; idx += 256) {
    int t = idx >> 6, k = idx & 63;
    float f = lst[(size_t)bh * Tc + t] * __expf(mst[(size_t)bh * Tc + t] - M) * invZ;
    attbf[((size_t)b * Tc + t) * Dc + h * DKc + k] =
        (bf16)(Ows[((size_t)bh * Tc + t) * DKc + k] * f);
  }
}

// ---------------- Kernel 5: output projection GEMM ----------------
__global__ void out_gemm(const bf16* __restrict__ attbf, const bf16* __restrict__ woT,
                         float* __restrict__ out) {
  int w = threadIdx.x >> 5, lane = threadIdx.x & 31;
  int mi = w & 3, ni = w >> 2;
  int m0 = blockIdx.x * 64 + mi * 16;
  int n0 = blockIdx.y * 32 + ni * 16;
  const bf16* Arow = attbf + (size_t)m0 * Dc;
  const bf16* Brow = woT + (size_t)n0 * Dc;
  v8f acc = {};
#pragma unroll 4
  for (int c = 0; c < Dc; c += 32) {
    v16bf a = load_a_frag(Arow + c, Dc, lane);
    v16bf b = load_b_frag(Brow + c, Dc, lane);
    acc = wmma_bf16(a, b, acc);
  }
  int half = lane >> 4, nc = lane & 15;
#pragma unroll
  for (int r = 0; r < 8; ++r) {
    int row = m0 + r + half * 8;
    out[(size_t)row * Dc + n0 + nc] = acc[r];
  }
}

// ---------------- host launcher ----------------
extern "C" void kernel_launch(void* const* d_in, const int* in_sizes, int n_in,
                              void* d_out, int out_size, void* d_ws, size_t ws_size,
                              hipStream_t stream) {
  const float* query = (const float*)d_in[0];
  const float* value = (const float*)d_in[1];
  const float* Wq = (const float*)d_in[2];
  const float* Wk = (const float*)d_in[3];
  const float* Wv = (const float*)d_in[4];
  const float* Wo = (const float*)d_in[5];
  const int* ls = (const int*)d_in[6];
  float* out = (float*)d_out;

  // workspace carve-up (256B aligned)
  char* ws = (char*)d_ws;
  size_t off = 0;
  auto carve = [&](size_t bytes) {
    void* p = ws + off;
    off = (off + bytes + 255) & ~(size_t)255;
    return p;
  };
  const size_t NTD = (size_t)NROW * Dc;        // 2,097,152
  bf16* qbf   = (bf16*)carve(NTD * 2);
  bf16* vbf   = (bf16*)carve(NTD * 2);
  bf16* wqkvT = (bf16*)carve((size_t)NQKV * Dc * 2);
  bf16* woT   = (bf16*)carve((size_t)Dc * Dc * 2);
  bf16* Qb    = (bf16*)carve(NTD * 2);
  bf16* Kb    = (bf16*)carve(NTD * 2);
  bf16* Vtb   = (bf16*)carve(NTD * 2);
  float* Ows  = (float*)carve(NTD * 4);
  float* mst  = (float*)carve((size_t)Bc * Hc * Tc * 4);
  float* lst  = (float*)carve((size_t)Bc * Hc * Tc * 4);
  bf16* attbf = (bf16*)carve(NTD * 2);
  (void)ws_size; (void)in_sizes; (void)n_in; (void)out_size;

  convert_pack<<<(NROW * Dc + 255) / 256, 256, 0, stream>>>(
      query, value, Wq, Wk, Wv, Wo, qbf, vbf, wqkvT, woT);

  proj_gemm<<<dim3(NROW / 64, NQKV / 32), 256, 0, stream>>>(
      qbf, vbf, wqkvT, Qb, Kb, Vtb);

  attn_flash<<<dim3(Tc / 128, Hc, Bc), 256, 0, stream>>>(
      Qb, Kb, Vtb, Ows, mst, lst, ls);

  stats_rescale<<<Bc * Hc, 256, 0, stream>>>(Ows, mst, lst, attbf);

  out_gemm<<<dim3(NROW / 64, Dc / 32), 256, 0, stream>>>(attbf, woT, out);
}
